// GroupedQueryAttention_30734785970876
// MI455X (gfx1250) — matrix operand order repaired and checked by reading, not compiled, not run on hardware
//
#include <hip/hip_runtime.h>
#include <hip/hip_bf16.h>

// ---------------------------------------------------------------------------
// GQA attention block for MI455X (gfx1250, wave32, WMMA).
// Pipeline: fp32 -> bf16 converts/transposes, bf16 WMMA GEMM projections,
// RoPE, flash-attention with online softmax, bf16 WMMA output projection.
// All matmuls use v_wmma_f32_16x16x32_bf16 (f32 accumulate).
// GEMM: 64x64 tile per wave (4x4 WMMA accumulators) for 1:1 wmma:b128-load.
// ---------------------------------------------------------------------------

typedef __attribute__((ext_vector_type(16))) __bf16 v16bf;
typedef __attribute__((ext_vector_type(8)))  float  v8f;

union Frag {
  v16bf v;
  unsigned int u[8];
};

__device__ __forceinline__ unsigned short f2bf(float f) {
  unsigned int u = __float_as_uint(f);
  unsigned int r = u + 0x7FFFu + ((u >> 16) & 1u);  // round-to-nearest-even
  return (unsigned short)(r >> 16);
}

// --------------------------- elementwise converts ---------------------------

__global__ void cvt_f32_bf16(const float* __restrict__ in,
                             unsigned short* __restrict__ out, long n) {
  long i = (long)blockIdx.x * blockDim.x + threadIdx.x;
  if (i < n) out[i] = f2bf(in[i]);
}

// in: (R x C) row-major f32  ->  out: (C x R) row-major bf16
__global__ void transpose_cvt_bf16(const float* __restrict__ in,
                                   unsigned short* __restrict__ out,
                                   int R, int C) {
  long i = (long)blockIdx.x * blockDim.x + threadIdx.x;
  long total = (long)R * C;
  if (i >= total) return;
  int k = (int)(i % R);   // fast index = row of input -> contiguous output
  int n = (int)(i / R);
  out[(long)n * R + k] = f2bf(in[(long)k * C + n]);
}

// --------------------------------- GEMM -------------------------------------
// C(f32, MxN row-major) = A(bf16, MxK row-major) * Bt(bf16, NxK row-major)^T
// One wave per 64x64 output tile: 4 M-subtiles x 4 N-subtiles of 16x16.
// Per 32-deep k-chunk: 4 A fragments + 4 B fragments (16x global_load_b128)
// feed 16 v_wmma_f32_16x16x32_bf16. M % 64 == 0, N % 64 == 0, K % 32 == 0.

__global__ void __launch_bounds__(32)
gemm_bf16_f32(const unsigned short* __restrict__ A,
              const unsigned short* __restrict__ Bt,
              float* __restrict__ C, int M, int N, int K) {
  const int lane = threadIdx.x & 31;
  const int l15  = lane & 15;
  const int half = (lane >> 4) & 1;
  const int mt = blockIdx.x;  // M/64
  const int nt = blockIdx.y;  // N/64

  const unsigned short* arow[4];
  const unsigned short* brow[4];
#pragma unroll
  for (int i = 0; i < 4; ++i)
    arow[i] = A + (long)(mt * 64 + i * 16 + l15) * K;
#pragma unroll
  for (int j = 0; j < 4; ++j)
    brow[j] = Bt + (long)(nt * 64 + j * 16 + l15) * K;

  v8f acc[4][4] = {};
  for (int kc = 0; kc < K; kc += 32) {
    Frag a[4];
#pragma unroll
    for (int i = 0; i < 4; ++i) {
      __builtin_prefetch(arow[i] + kc + 32, 0, 1);  // global_prefetch_b8
#pragma unroll
      for (int v = 0; v < 8; ++v) {
        int k = kc + ((v >= 4) ? 16 : 0) + (half ? 8 : 0) + (v & 3) * 2;
        a[i].u[v] = *(const unsigned int*)(arow[i] + k);
      }
    }
#pragma unroll
    for (int j = 0; j < 4; ++j) {
      Frag b;
#pragma unroll
      for (int v = 0; v < 8; ++v)
        b.u[v] = *(const unsigned int*)(brow[j] + kc + ((v >= 4) ? 16 : 0) + (half ? 8 : 0) + (v & 3) * 2);
#pragma unroll
      for (int i = 0; i < 4; ++i)
        acc[i][j] = __builtin_amdgcn_wmma_f32_16x16x32_bf16(
            false, a[i].v, false, b.v, (short)0, acc[i][j], false, false);
    }
  }
#pragma unroll
  for (int i = 0; i < 4; ++i)
#pragma unroll
    for (int j = 0; j < 4; ++j) {
      long ncol = (long)nt * 64 + j * 16 + l15;
#pragma unroll
      for (int v = 0; v < 8; ++v) {
        long r = (long)mt * 64 + i * 16 + v + (half ? 8 : 0);
        C[r * N + ncol] = acc[i][j][v];
      }
    }
}

// --------------------------------- RoPE -------------------------------------
// P: (B*T, H*128) f32 projection.  Out: (B,H,T,128) bf16 with RoPE applied,
// optionally pre-scaled (Q gets 1/sqrt(128)).

__global__ void rope_to_bhtd(const float* __restrict__ P,
                             unsigned short* __restrict__ Out,
                             int Bn, int T, int H, float scale) {
  long idx = (long)blockIdx.x * blockDim.x + threadIdx.x;
  long total = (long)Bn * H * T * 64;
  if (idx >= total) return;
  int i = (int)(idx & 63);
  long tmp = idx >> 6;
  int t = (int)(tmp % T); tmp /= T;
  int h = (int)(tmp % H); tmp /= H;
  int b = (int)tmp;

  const float* p = P + ((long)(b * T + t) * H + h) * 128;
  float inv = powf(10000.0f, -(float)i / 64.0f);
  float ang = (float)t * inv;
  float c = cosf(ang), s = sinf(ang);
  float x0 = p[i], x1 = p[i + 64];
  unsigned short* o = Out + ((long)(b * H + h) * T + t) * 128;
  o[i]      = f2bf((x0 * c - x1 * s) * scale);
  o[i + 64] = f2bf((x1 * c + x0 * s) * scale);
}

// V: (B*T, 4*128) f32 -> Vt: (B,4,128,T) bf16 (transposed for B-fragments)
__global__ void v_to_transposed(const float* __restrict__ P,
                                unsigned short* __restrict__ Vt,
                                int Bn, int T) {
  long idx = (long)blockIdx.x * blockDim.x + threadIdx.x;
  long total = (long)Bn * 4 * 128 * T;
  if (idx >= total) return;
  int t = (int)(idx % T);
  long tmp = idx / T;
  int d = (int)(tmp % 128); tmp /= 128;
  int h = (int)(tmp % 4);   tmp /= 4;
  int b = (int)tmp;
  Vt[idx] = f2bf(P[((long)(b * T + t) * 4 + h) * 128 + d]);
}

// ----------------------------- flash attention ------------------------------
// One wave per 16-query tile of one (b, head).  Online softmax over 32-key
// chunks.  S = Q K^T via WMMA (C layout: lane=key col, VGPR=query row);
// stats reduced with shfl_xor inside 16-lane halves; P bounced through LDS
// to re-layout as an A fragment for P @ V.

__global__ void __launch_bounds__(32)
flash_attn(const unsigned short* __restrict__ Q,   // (B,16,T,128) bf16 (scaled)
           const unsigned short* __restrict__ Kb,  // (B, 4,T,128) bf16
           const unsigned short* __restrict__ Vt,  // (B, 4,128,T) bf16
           unsigned short* __restrict__ Y,         // (B,T,2048) bf16
           int Bn, int T) {
  __shared__ unsigned short pbuf[16 * 32];
  (void)Bn;

  const int lane = threadIdx.x & 31;
  const int l15  = lane & 15;
  const int half = (lane >> 4) & 1;

  const int ntq = T / 16;
  int qt = blockIdx.x;
  const int t16 = qt % ntq;
  const int h   = (qt / ntq) % 16;
  const int b   = qt / (ntq * 16);
  const int hkv = h >> 2;  // N_REP = 4
  const int qbase = t16 * 16;

  const unsigned short* Qh = Q  + (long)(b * 16 + h)  * T * 128;
  const unsigned short* Kh = Kb + (long)(b * 4 + hkv) * T * 128;
  const unsigned short* Vh = Vt + (long)(b * 4 + hkv) * 128 * T;

  // Q tile 16x128 as four 16x32 A fragments
  Frag qf[4];
  {
    const unsigned short* qrow = Qh + (long)(qbase + l15) * 128;
#pragma unroll
    for (int c = 0; c < 4; ++c)
#pragma unroll
      for (int v = 0; v < 8; ++v) {
        int k = c * 32 + ((v >= 4) ? 16 : 0) + (half ? 8 : 0) + (v & 3) * 2;
        qf[c].u[v] = *(const unsigned int*)(qrow + k);
      }
  }

  v8f o[8] = {};              // 16x128 f32 accumulator
  float mrow[8], lrow[8];
#pragma unroll
  for (int v = 0; v < 8; ++v) { mrow[v] = -1e30f; lrow[v] = 0.0f; }

  const int nchunks = (qbase + 15) / 32 + 1;
  for (int ch = 0; ch < nchunks; ++ch) {
    const int kbase = ch * 32;

    // S = Q K^T for two 16-key tiles
    v8f s[2] = {};
#pragma unroll
    for (int t = 0; t < 2; ++t) {
      const unsigned short* krow =
          Kh + (long)(kbase + t * 16 + l15) * 128;
#pragma unroll
      for (int c = 0; c < 4; ++c) {
        Frag kf;
#pragma unroll
        for (int v = 0; v < 8; ++v)
          kf.u[v] = *(const unsigned int*)(krow + c * 32 + ((v >= 4) ? 16 : 0) + (half ? 8 : 0) + (v & 3) * 2);
        s[t] = __builtin_amdgcn_wmma_f32_16x16x32_bf16(
            false, qf[c].v, false, kf.v, (short)0, s[t], false, false);
      }
    }

    // causal mask + per-row chunk max
    float cmax[8];
#pragma unroll
    for (int v = 0; v < 8; ++v) {
      int qi = qbase + v + (half ? 8 : 0);
      float s0 = (kbase + l15      <= qi) ? s[0][v] : -1e30f;
      float s1 = (kbase + 16 + l15 <= qi) ? s[1][v] : -1e30f;
      s[0][v] = s0; s[1][v] = s1;
      cmax[v] = fmaxf(s0, s1);
    }
#pragma unroll
    for (int d = 1; d < 16; d <<= 1)
#pragma unroll
      for (int v = 0; v < 8; ++v)
        cmax[v] = fmaxf(cmax[v], __shfl_xor(cmax[v], d, 32));

    // online softmax update; write P (bf16) to LDS in row-major (m, k)
    float alpha[8], csum[8];
#pragma unroll
    for (int v = 0; v < 8; ++v) {
      float mn = fmaxf(mrow[v], cmax[v]);
      alpha[v] = __expf(mrow[v] - mn);
      mrow[v] = mn;
      float p0 = __expf(s[0][v] - mn);
      float p1 = __expf(s[1][v] - mn);
      csum[v] = p0 + p1;
      int m = v + (half ? 8 : 0);
      pbuf[m * 32 + l15]      = f2bf(p0);
      pbuf[m * 32 + 16 + l15] = f2bf(p1);
    }
#pragma unroll
    for (int d = 1; d < 16; d <<= 1)
#pragma unroll
      for (int v = 0; v < 8; ++v)
        csum[v] += __shfl_xor(csum[v], d, 32);
#pragma unroll
    for (int v = 0; v < 8; ++v) lrow[v] = lrow[v] * alpha[v] + csum[v];
#pragma unroll
    for (int nt = 0; nt < 8; ++nt)
#pragma unroll
      for (int v = 0; v < 8; ++v) o[nt][v] *= alpha[v];

    __syncthreads();

    // reload P as a 16x32 A fragment
    Frag pf;
#pragma unroll
    for (int v = 0; v < 8; ++v) {
      int k = ((v >= 4) ? 16 : 0) + (half ? 8 : 0) + (v & 3) * 2;
      pf.u[v] = *(const unsigned int*)(&pbuf[l15 * 32 + k]);
    }

    // O += P @ V over eight 16-wide dim tiles
#pragma unroll
    for (int nt = 0; nt < 8; ++nt) {
      Frag vf;
      const unsigned short* vrow =
          Vh + (long)(nt * 16 + l15) * T + kbase;
#pragma unroll
      for (int v = 0; v < 8; ++v)
        vf.u[v] = *(const unsigned int*)(vrow + ((v >= 4) ? 16 : 0) + (half ? 8 : 0) + (v & 3) * 2);
      o[nt] = __builtin_amdgcn_wmma_f32_16x16x32_bf16(
          false, pf.v, false, vf.v, (short)0, o[nt], false, false);
    }

    __syncthreads();
  }

  // normalize and emit y as (B,T,H*128) bf16
#pragma unroll
  for (int nt = 0; nt < 8; ++nt)
#pragma unroll
    for (int v = 0; v < 8; ++v) {
      int qi = qbase + v + (half ? 8 : 0);
      float val = o[nt][v] / lrow[v];
      Y[(long)(b * T + qi) * 2048 + h * 128 + nt * 16 + l15] = f2bf(val);
    }
}

// ------------------------------- launch -------------------------------------

extern "C" void kernel_launch(void* const* d_in, const int* in_sizes, int n_in,
                              void* d_out, int out_size, void* d_ws, size_t ws_size,
                              hipStream_t stream) {
  (void)in_sizes; (void)n_in; (void)out_size; (void)ws_size;

  constexpr int B = 2, T = 2048, D = 2048, H = 16, HKV = 4, HD = 128;
  constexpr int M  = B * T;       // 4096
  constexpr int KV = HKV * HD;    // 512

  const float* x  = (const float*)d_in[0];
  const float* Wq = (const float*)d_in[1];
  const float* Wk = (const float*)d_in[2];
  const float* Wv = (const float*)d_in[3];
  const float* Wo = (const float*)d_in[4];
  float* out = (float*)d_out;

  char* ws = (char*)d_ws;
  size_t off = 0;
  auto alloc = [&](size_t bytes) -> void* {
    void* p = ws + off;
    off += (bytes + 255) & ~(size_t)255;
    return p;
  };
  unsigned short* xb  = (unsigned short*)alloc((size_t)M * D * 2);
  unsigned short* wqt = (unsigned short*)alloc((size_t)D * D * 2);
  unsigned short* wkt = (unsigned short*)alloc((size_t)KV * D * 2);
  unsigned short* wvt = (unsigned short*)alloc((size_t)KV * D * 2);
  unsigned short* wot = (unsigned short*)alloc((size_t)D * D * 2);
  float*          pf  = (float*)alloc((size_t)M * D * 4);   // reused: Q, then K, then V proj
  unsigned short* qb  = (unsigned short*)alloc((size_t)M * D * 2);
  unsigned short* kb  = (unsigned short*)alloc((size_t)M * KV * 2);
  unsigned short* vt  = (unsigned short*)alloc((size_t)M * KV * 2);
  unsigned short* yb  = (unsigned short*)alloc((size_t)M * D * 2);

  auto blks = [](long n) { return (unsigned)((n + 255) / 256); };

  // bf16 conversions / weight transposes
  cvt_f32_bf16<<<blks((long)M * D), 256, 0, stream>>>(x, xb, (long)M * D);
  transpose_cvt_bf16<<<blks((long)D * D), 256, 0, stream>>>(Wq, wqt, D, D);
  transpose_cvt_bf16<<<blks((long)D * KV), 256, 0, stream>>>(Wk, wkt, D, KV);
  transpose_cvt_bf16<<<blks((long)D * KV), 256, 0, stream>>>(Wv, wvt, D, KV);
  transpose_cvt_bf16<<<blks((long)D * D), 256, 0, stream>>>(Wo, wot, D, D);

  const float qscale = 0.088388347648318447f;  // 1/sqrt(128)

  // Q projection + RoPE (scale folded into Q)
  gemm_bf16_f32<<<dim3(M / 64, D / 64), 32, 0, stream>>>(xb, wqt, pf, M, D, D);
  rope_to_bhtd<<<blks((long)B * H * T * 64), 256, 0, stream>>>(pf, qb, B, T, H, qscale);

  // K projection + RoPE
  gemm_bf16_f32<<<dim3(M / 64, KV / 64), 32, 0, stream>>>(xb, wkt, pf, M, KV, D);
  rope_to_bhtd<<<blks((long)B * HKV * T * 64), 256, 0, stream>>>(pf, kb, B, T, HKV, 1.0f);

  // V projection + transpose
  gemm_bf16_f32<<<dim3(M / 64, KV / 64), 32, 0, stream>>>(xb, wvt, pf, M, KV, D);
  v_to_transposed<<<blks((long)B * HKV * HD * T), 256, 0, stream>>>(pf, vt, B, T);

  // grouped causal attention
  flash_attn<<<B * H * (T / 16), 32, 0, stream>>>(qb, kb, vt, yb, B, T);

  // output projection -> f32 result
  gemm_bf16_f32<<<dim3(M / 64, D / 64), 32, 0, stream>>>(yb, wot, out, M, D, D);
}
